// GATConv_56908316672628
// MI455X (gfx1250) — matrix operand order repaired and checked by reading, not compiled, not run on hardware
//
#include <hip/hip_runtime.h>
#include <hip/hip_bf16.h>

typedef __bf16 v16bf __attribute__((ext_vector_type(16)));
typedef float  v8f   __attribute__((ext_vector_type(8)));

#define DIN 128
#define DOUT 64
#define NEG_SLOPE 0.2f

// ---------- helpers ----------
__device__ __forceinline__ unsigned ord_f32(float f) {
    unsigned u = __float_as_uint(f);
    return (u & 0x80000000u) ? ~u : (u | 0x80000000u);
}
__device__ __forceinline__ float unord_f32(unsigned u) {
    return (u & 0x80000000u) ? __uint_as_float(u & 0x7fffffffu)
                             : __uint_as_float(~u);
}
__device__ __forceinline__ float leaky(float v) {
    return v >= 0.0f ? v : NEG_SLOPE * v;
}

// ---------- K0: init scratch + zero output ----------
__global__ void k_init(float* __restrict__ out, unsigned* __restrict__ emax,
                       float* __restrict__ esum, int N) {
    long long i = (long long)blockIdx.x * blockDim.x + threadIdx.x;
    long long tot = (long long)N * DOUT;
    if (i < tot) out[i] = 0.0f;
    if (i < N) { emax[i] = 0u; esum[i] = 0.0f; }   // 0 < ord(-inf)
}

// ---------- K1: h = x @ W via v_wmma_f32_16x16x32_bf16 ----------
// block = 256 threads = 8 waves, each wave -> 16 rows x 64 cols
#define WT_STRIDE 136  // pad 128 -> 136 bf16 to spread LDS banks
__global__ __launch_bounds__(256)
void k_gemm_wmma(const float* __restrict__ x, const float* __restrict__ W,
                 float* __restrict__ h, int N) {
    __shared__ __bf16 Wt[DOUT * WT_STRIDE];  // W transposed: Wt[n][k], bf16 (~17 KB)

    int tid = threadIdx.x;
    for (int i = tid; i < DIN * DOUT; i += 256) {
        int k = i >> 6, n = i & 63;                 // W row-major [k][n]
        Wt[n * WT_STRIDE + k] = (__bf16)W[i];
    }
    __syncthreads();

    const int wave = tid >> 5;
    const int lane = tid & 31;
    const int half = lane >> 4;      // 0: lanes 0-15, 1: lanes 16-31
    const int rl   = lane & 15;

    const int row0 = blockIdx.x * 128 + wave * 16;
    int arow = row0 + rl;
    if (arow >= N) arow = N - 1;     // clamp loads; stores are guarded
    const float* xrow = x + (size_t)arow * DIN;

    v8f acc0 = {}, acc1 = {}, acc2 = {}, acc3 = {};

#pragma unroll
    for (int ks = 0; ks < DIN; ks += 32) {
        // A operand: 16x32 bf16. lane half selects K sub-ranges (contiguous, matches B):
        // elems 0..15 -> K = ks + half*16 + j
        v16bf a;
        const int k0 = ks + half * 16;
#pragma unroll
        for (int j = 0; j < 8; ++j) {
            a[j]     = (__bf16)xrow[k0 + j];
            a[8 + j] = (__bf16)xrow[k0 + 8 + j];
        }
        // B operands: column n = nt*16 + rl, K = ks + half*16 + (0..15).
        // Load all 4 tiles first so the 4 WMMAs issue back-to-back after one
        // dscnt wait, instead of a load->wait->wmma serial chain.
        const __bf16* wb = &Wt[rl * WT_STRIDE + ks + half * 16];
        v16bf b0, b1, b2, b3;
#pragma unroll
        for (int j = 0; j < 16; ++j) b0[j] = wb[j];
#pragma unroll
        for (int j = 0; j < 16; ++j) b1[j] = wb[16 * WT_STRIDE + j];
#pragma unroll
        for (int j = 0; j < 16; ++j) b2[j] = wb[32 * WT_STRIDE + j];
#pragma unroll
        for (int j = 0; j < 16; ++j) b3[j] = wb[48 * WT_STRIDE + j];

        acc0 = __builtin_amdgcn_wmma_f32_16x16x32_bf16(false, a, false, b0,
                                                       (short)0, acc0, false, false);
        acc1 = __builtin_amdgcn_wmma_f32_16x16x32_bf16(false, a, false, b1,
                                                       (short)0, acc1, false, false);
        acc2 = __builtin_amdgcn_wmma_f32_16x16x32_bf16(false, a, false, b2,
                                                       (short)0, acc2, false, false);
        acc3 = __builtin_amdgcn_wmma_f32_16x16x32_bf16(false, a, false, b3,
                                                       (short)0, acc3, false, false);
    }

    // C/D layout: vgpr v, lanes0-15 -> M = v ; lanes16-31 -> M = 8+v ; col = lane%16
#pragma unroll
    for (int v = 0; v < 8; ++v) {
        const int r = row0 + half * 8 + v;
        if (r < N) {
            float* hr = h + (size_t)r * DOUT + rl;
            hr[0]  = acc0[v];
            hr[16] = acc1[v];
            hr[32] = acc2[v];
            hr[48] = acc3[v];
        }
    }
}

// ---------- K1b: per-node attention scores (warp per node) ----------
__global__ __launch_bounds__(256)
void k_scores(const float* __restrict__ h, const float* __restrict__ att_src,
              const float* __restrict__ att_dst, float* __restrict__ a_src,
              float* __restrict__ a_dst, int N) {
    const int lane = threadIdx.x & 31;
    const int node = blockIdx.x * 8 + (threadIdx.x >> 5);
    if (node >= N) return;
    const float* hr = h + (size_t)node * DOUT;
    float h0 = hr[lane], h1 = hr[lane + 32];
    float s = h0 * att_src[lane] + h1 * att_src[lane + 32];
    float d = h0 * att_dst[lane] + h1 * att_dst[lane + 32];
#pragma unroll
    for (int m = 16; m >= 1; m >>= 1) {
        s += __shfl_xor(s, m, 32);
        d += __shfl_xor(d, m, 32);
    }
    if (lane == 0) { a_src[node] = s; a_dst[node] = d; }
}

// ---------- K2: segment max of leaky(a_src[s]+a_dst[d]) over dst ----------
__global__ void k_edge_max(const int* __restrict__ ei, const float* __restrict__ a_src,
                           const float* __restrict__ a_dst, unsigned* __restrict__ emax,
                           int E, int N) {
    long long i = (long long)blockIdx.x * blockDim.x + threadIdx.x;
    long long Etot = (long long)E + N;
    if (i >= Etot) return;
    int s, d;
    if (i < E) { s = ei[i]; d = ei[(long long)E + i]; }
    else       { s = (int)(i - E); d = s; }          // self loop
    float e = leaky(a_src[s] + a_dst[d]);
    atomicMax(&emax[d], ord_f32(e));
}

// ---------- K3: fused exp + weighted scatter-add (wave per edge) ----------
__global__ __launch_bounds__(256)
void k_edge_scatter(const int* __restrict__ ei, const float* __restrict__ a_src,
                    const float* __restrict__ a_dst, const unsigned* __restrict__ emax,
                    const float* __restrict__ h, float* __restrict__ out,
                    float* __restrict__ esum, int E, int N) {
    const int lane = threadIdx.x & 31;
    long long edge = (long long)blockIdx.x * 8 + (threadIdx.x >> 5);
    long long Etot = (long long)E + N;
    if (edge >= Etot) return;
    int s, d;
    if (edge < E) { s = ei[edge]; d = ei[(long long)E + edge]; }
    else          { s = (int)(edge - E); d = s; }
    float e = leaky(a_src[s] + a_dst[d]);
    float w = __expf(e - unord_f32(emax[d]));
    const float* hs = h + (size_t)s * DOUT;
    float* od = out + (size_t)d * DOUT;
    unsafeAtomicAdd(&od[lane],      hs[lane]      * w);
    unsafeAtomicAdd(&od[lane + 32], hs[lane + 32] * w);
    if (lane == 0) unsafeAtomicAdd(&esum[d], w);
}

// ---------- K4: normalize + bias ----------
__global__ void k_finalize(float* __restrict__ out, const float* __restrict__ esum,
                           const float* __restrict__ bias, int N) {
    long long i = (long long)blockIdx.x * blockDim.x + threadIdx.x;
    if (i >= (long long)N * DOUT) return;
    int n = (int)(i >> 6), c = (int)(i & 63);
    out[i] = out[i] / (esum[n] + 1e-16f) + bias[c];
}

extern "C" void kernel_launch(void* const* d_in, const int* in_sizes, int n_in,
                              void* d_out, int out_size, void* d_ws, size_t ws_size,
                              hipStream_t stream) {
    const float* x       = (const float*)d_in[0];
    const int*   ei      = (const int*)d_in[1];
    const float* W       = (const float*)d_in[2];
    const float* att_src = (const float*)d_in[3];
    const float* att_dst = (const float*)d_in[4];
    const float* bias    = (const float*)d_in[5];
    float* out = (float*)d_out;

    const int N = in_sizes[0] / DIN;      // 100000
    const int E = in_sizes[1] / 2;        // 1600000
    const long long Etot = (long long)E + N;

    // carve workspace (256B-aligned chunks)
    char* p = (char*)d_ws;
    auto carve = [&](size_t bytes) { char* q = p; p += (bytes + 255) & ~size_t(255); return q; };
    float*    h     = (float*)   carve((size_t)N * DOUT * sizeof(float));
    float*    a_src = (float*)   carve((size_t)N * sizeof(float));
    float*    a_dst = (float*)   carve((size_t)N * sizeof(float));
    unsigned* emax  = (unsigned*)carve((size_t)N * sizeof(unsigned));
    float*    esum  = (float*)   carve((size_t)N * sizeof(float));

    const long long NC = (long long)N * DOUT;
    k_init<<<(int)((NC + 255) / 256), 256, 0, stream>>>(out, emax, esum, N);

    k_gemm_wmma<<<(N + 127) / 128, 256, 0, stream>>>(x, W, h, N);

    k_scores<<<(N + 7) / 8, 256, 0, stream>>>(h, att_src, att_dst, a_src, a_dst, N);

    k_edge_max<<<(int)((Etot + 255) / 256), 256, 0, stream>>>(ei, a_src, a_dst, emax, E, N);

    k_edge_scatter<<<(int)((Etot + 7) / 8), 256, 0, stream>>>(ei, a_src, a_dst, emax,
                                                              h, out, esum, E, N);

    k_finalize<<<(int)((NC + 255) / 256), 256, 0, stream>>>(out, esum, bias, N);
}
